// IPFECNN_22582938042585
// MI455X (gfx1250) — matrix operand order repaired and checked by reading, not compiled, not run on hardware
//
#include <hip/hip_runtime.h>
#include <hip/hip_bf16.h>

typedef __attribute__((ext_vector_type(16))) _Float16 v16h;
typedef __attribute__((ext_vector_type(8)))  _Float16 v8h;
typedef __attribute__((ext_vector_type(8)))  float    v8f;
typedef __attribute__((ext_vector_type(4)))  unsigned int v4u;

#define EPSF 1e-5f

// workspace layout (bytes), all 16B aligned; feat1/feat2 have 1-px zero halos
#define FEAT1P_OFF 0ull          // f16 [1024][16][16][16] = 8,388,608 B (halo)
#define FEAT2P_OFF 8388608ull    // f16 [1024][9][9][32]   = 5,308,416 B (halo)
#define FEAT3_OFF  13697024ull   // f16 [1024][576]        = 1,179,648 B
#define WF1S_OFF   14876672ull   // f16 [18][128][32]      =   147,456 B (WMMA-B packed)
#define HBUF_OFF   15024128ull   // f16 [1024][128]        =   262,144 B
#define ZERO_N16   856064        // (FEAT1P+FEAT2P bytes)/16

// ---------------------------------------------------------------------------
// zero-fill the two halo'd activation buffers (16B vector stores)
// ---------------------------------------------------------------------------
__global__ void k_zero(v4u* __restrict__ p, int n16)
{
    int i = blockIdx.x * blockDim.x + threadIdx.x;
    if (i < n16) p[i] = (v4u){};
}

// ---------------------------------------------------------------------------
// conv1: exact int32 IPFE dot, /1e4+bias folded into BN affine, ReLU, 2x2 pool
// one thread per (b,oy,ox), all 16 channels: 16 pixel loads reused 16x,
// weights/BN params are thread-uniform -> scalar loads. Output into halo'd
// NHWC buffer [1024][16][16][16], interior at (+1,+1).
// ---------------------------------------------------------------------------
__global__ void k_conv1_bn_pool(const int* __restrict__ x, const int* __restrict__ yw,
                                const float* __restrict__ bias1,
                                const float* __restrict__ gg, const float* __restrict__ bb,
                                const float* __restrict__ mm, const float* __restrict__ vv,
                                _Float16* __restrict__ feat1p)
{
    int idx = blockIdx.x * blockDim.x + threadIdx.x;
    if (idx >= 1024 * 196) return;
    int ox = idx % 14; int t = idx / 14;
    int oy = t % 14;   int b = t / 14;

    // 4x4 input patch covering the 2x2 pool window's 3x3 taps
    int px[4][4];
    const int* xb = x + b * 784;
    int y0 = 2 * oy - 1, x0 = 2 * ox - 1;
#pragma unroll
    for (int iy = 0; iy < 4; ++iy) {
        int yy = y0 + iy;
        bool yok = (yy >= 0) && (yy < 28);
#pragma unroll
        for (int ix = 0; ix < 4; ++ix) {
            int xx = x0 + ix;
            px[iy][ix] = (yok && xx >= 0 && xx < 28) ? xb[yy * 28 + xx] : 0;
        }
    }

    _Float16 outv[16];
#pragma unroll
    for (int c = 0; c < 16; ++c) {
        float inv = gg[c] * rsqrtf(vv[c] + EPSF);
        float sc  = inv * 1e-4f;
        float tc  = bias1[c] * inv + bb[c] - mm[c] * inv;
        float best = -1e30f;
#pragma unroll
        for (int py = 0; py < 2; ++py)
#pragma unroll
        for (int qx = 0; qx < 2; ++qx) {
            int acc = 0;
#pragma unroll
            for (int dy = 0; dy < 3; ++dy)
#pragma unroll
            for (int dx = 0; dx < 3; ++dx)
                acc += px[py + dy][qx + dx] * yw[c * 9 + dy * 3 + dx];
            best = fmaxf(best, (float)acc * sc + tc);
        }
        outv[c] = (_Float16)fmaxf(best, 0.f);
    }

    v8h* dp = (v8h*)(feat1p + ((b * 16 + oy + 1) * 16 + (ox + 1)) * 16);
    v8h o0, o1;
#pragma unroll
    for (int i = 0; i < 8; ++i) { o0[i] = outv[i]; o1[i] = outv[i + 8]; }
    dp[0] = o0;
    dp[1] = o1;
}

// ---------------------------------------------------------------------------
// conv2 (implicit GEMM, WMMA f16): M=16 images, N=16 of 32 out-ch, K=144 pad 160
// halo'd input -> unconditional global_load_b128 with immediate tap offsets.
// each wave owns a 2x2 pooling quad (4 accumulators); fused BN+ReLU+pool.
// ---------------------------------------------------------------------------
__global__ void k_conv2_bn_pool(const _Float16* __restrict__ feat1p, const float* __restrict__ w2,
                                const float* __restrict__ bias2,
                                const float* __restrict__ gg, const float* __restrict__ bb,
                                const float* __restrict__ mm, const float* __restrict__ vv,
                                _Float16* __restrict__ feat2p)
{
    __shared__ __align__(32) _Float16 Bs[32][160];   // [N][K], k = tap*16 + c_in
    for (int idx = threadIdx.x; idx < 32 * 160; idx += blockDim.x) {
        int n = idx / 160, k = idx % 160;
        _Float16 val = (_Float16)0.f;
        if (k < 144) { int tap = k >> 4, c = k & 15; val = (_Float16)w2[n * 144 + c * 9 + tap]; }
        Bs[n][k] = val;
    }
    __syncthreads();

    int wave = (blockIdx.x * blockDim.x + threadIdx.x) >> 5;
    int lane = threadIdx.x & 31;
    int ntile = wave & 1;
    int t = wave >> 1;
    int quad = t % 49;
    int imgt = t / 49;                 // 64 image tiles
    int qy = quad / 7, qx = quad % 7;
    int half = lane >> 4, l16 = lane & 15;
    int bimg = imgt * 16 + l16;

    v8f acc[4] = {};

#pragma unroll
    for (int ks = 0; ks < 5; ++ks) {   // K = 160 = 5 x 32 (2 taps per step)
        const _Float16* bp = &Bs[ntile * 16 + l16][ks * 32 + half * 8];
        v8h blo = *(const v8h*)bp;
        v8h bhi = *(const v8h*)(bp + 16);
        v16h bm;
#pragma unroll
        for (int i = 0; i < 8; ++i) { bm[i] = blo[i]; bm[i + 8] = bhi[i]; }
        const int t0 = 2 * ks, t1 = 2 * ks + 1;
        const int dy0 = t0 / 3 - 1, dx0 = t0 % 3 - 1;
#pragma unroll
        for (int p = 0; p < 4; ++p) {
            int y  = qy * 2 + (p >> 1) + 1;   // padded coords
            int xo = qx * 2 + (p & 1) + 1;
            const _Float16* ap = feat1p + ((bimg * 16 + y) * 16 + xo) * 16 + half * 8;
            v8h alo = *(const v8h*)(ap + (dy0 * 16 + dx0) * 16);
            v8h ahi = {};
            if (t1 < 9) {                      // compile-time after unroll
                const int dy1 = t1 / 3 - 1, dx1 = t1 % 3 - 1;
                ahi = *(const v8h*)(ap + (dy1 * 16 + dx1) * 16);
            }
            v16h am;
#pragma unroll
            for (int i = 0; i < 8; ++i) { am[i] = alo[i]; am[i + 8] = ahi[i]; }
            acc[p] = __builtin_amdgcn_wmma_f32_16x16x32_f16(false, am, false, bm,
                                                            (short)0, acc[p], false, false);
        }
    }

    int n = ntile * 16 + l16;
    float inv = gg[n] * rsqrtf(vv[n] + EPSF);
    float tc  = bias2[n] * inv + bb[n] - mm[n] * inv;
#pragma unroll
    for (int r = 0; r < 8; ++r) {
        float p0 = acc[0][r] * inv + tc, p1 = acc[1][r] * inv + tc;
        float p2 = acc[2][r] * inv + tc, p3 = acc[3][r] * inv + tc;
        float mx = fmaxf(fmaxf(fmaxf(p0, p1), fmaxf(p2, p3)), 0.f);
        int bo = imgt * 16 + r + 8 * half;
        feat2p[((bo * 9 + qy + 1) * 9 + qx + 1) * 32 + n] = (_Float16)mx;
    }
}

// ---------------------------------------------------------------------------
// conv3 (implicit GEMM, WMMA f16): K = 9 taps x 32 ch = 288 -> 9 K32 steps.
// halo'd input -> unconditional loads. Output in FC-flat CHW order.
// ---------------------------------------------------------------------------
__global__ void k_conv3_bn_pool(const _Float16* __restrict__ feat2p, const float* __restrict__ w3,
                                const float* __restrict__ bias3,
                                const float* __restrict__ gg, const float* __restrict__ bb,
                                const float* __restrict__ mm, const float* __restrict__ vv,
                                _Float16* __restrict__ feat3)
{
    __shared__ __align__(32) _Float16 Bs[64][288];   // [N][K], k = tap*32 + c_in (36 KB)
    for (int idx = threadIdx.x; idx < 64 * 288; idx += blockDim.x) {
        int n = idx / 288, k = idx % 288;
        int tap = k / 32, c = k % 32;
        Bs[n][k] = (_Float16)w3[n * 288 + c * 9 + tap];
    }
    __syncthreads();

    int wave = (blockIdx.x * blockDim.x + threadIdx.x) >> 5;
    int lane = threadIdx.x & 31;
    int ntile = wave & 3;              // 4 N-tiles of 16 (64 out channels)
    int t = wave >> 2;
    int quad = t % 9;                  // 3x3 pooled output
    int imgt = t / 9;                  // 64 image tiles
    int qy = quad / 3, qx = quad % 3;
    int half = lane >> 4, l16 = lane & 15;
    int bimg = imgt * 16 + l16;

    v8f acc[4] = {};

#pragma unroll
    for (int ks = 0; ks < 9; ++ks) {   // one tap per K32 step
        const _Float16* bp = &Bs[ntile * 16 + l16][ks * 32 + half * 8];
        v8h blo = *(const v8h*)bp;
        v8h bhi = *(const v8h*)(bp + 16);
        v16h bm;
#pragma unroll
        for (int i = 0; i < 8; ++i) { bm[i] = blo[i]; bm[i + 8] = bhi[i]; }
        const int dy = ks / 3 - 1, dx = ks % 3 - 1;
#pragma unroll
        for (int p = 0; p < 4; ++p) {
            int y  = qy * 2 + (p >> 1) + 1;   // padded coords
            int xo = qx * 2 + (p & 1) + 1;
            const _Float16* ap = feat2p + ((bimg * 9 + y + dy) * 9 + (xo + dx)) * 32;
            v8h alo = *(const v8h*)(ap + half * 8);        // K = half*8 + i
            v8h ahi = *(const v8h*)(ap + 16 + half * 8);   // K = 16 + half*8 + i
            v16h am;
#pragma unroll
            for (int i = 0; i < 8; ++i) { am[i] = alo[i]; am[i + 8] = ahi[i]; }
            acc[p] = __builtin_amdgcn_wmma_f32_16x16x32_f16(false, am, false, bm,
                                                            (short)0, acc[p], false, false);
        }
    }

    int n = ntile * 16 + l16;
    float inv = gg[n] * rsqrtf(vv[n] + EPSF);
    float tc  = bias3[n] * inv + bb[n] - mm[n] * inv;
#pragma unroll
    for (int r = 0; r < 8; ++r) {
        float p0 = acc[0][r] * inv + tc, p1 = acc[1][r] * inv + tc;
        float p2 = acc[2][r] * inv + tc, p3 = acc[3][r] * inv + tc;
        float mx = fmaxf(fmaxf(fmaxf(p0, p1), fmaxf(p2, p3)), 0.f);
        int bo = imgt * 16 + r + 8 * half;
        feat3[bo * 576 + n * 9 + qy * 3 + qx] = (_Float16)mx;   // CHW flatten order
    }
}

// ---------------------------------------------------------------------------
// pack w_fc1 [128][576] f32 into WMMA-B layout f16: wf1s[ks][n][kk], kk=0..31
// -> each lane reads two 16 B runs (kk = 8h+i and 16+8h+i) per K-step
// ---------------------------------------------------------------------------
__global__ void k_wcvt(const float* __restrict__ wfc1, _Float16* __restrict__ wf1s)
{
    int idx = blockIdx.x * blockDim.x + threadIdx.x;
    if (idx >= 18 * 128 * 32) return;
    int kk = idx & 31;
    int n  = (idx >> 5) & 127;
    int ks = idx >> 12;
    wf1s[idx] = (_Float16)wfc1[n * 576 + ks * 32 + kk];
}

// ---------------------------------------------------------------------------
// FC1 (WMMA f16): [1024 x 576] x [576 x 128], K = 18 x 32 fully unrolled, ReLU
// ---------------------------------------------------------------------------
__global__ void k_fc1(const _Float16* __restrict__ feat3, const _Float16* __restrict__ wf1s,
                      const float* __restrict__ bfc1, _Float16* __restrict__ h)
{
    int wave = (blockIdx.x * blockDim.x + threadIdx.x) >> 5;
    int lane = threadIdx.x & 31;
    int ntile = wave & 7;              // 8 N-tiles (128 out)
    int mtile = wave >> 3;             // 64 M-tiles (1024 batch)
    int half = lane >> 4, l16 = lane & 15;
    int bimg = mtile * 16 + l16;
    int n = ntile * 16 + l16;

    const _Float16* abase = feat3 + bimg * 576 + half * 8;
    const _Float16* bbase = wf1s + n * 32 + half * 8;
    __builtin_prefetch(bbase, 0, 1);                  // global_prefetch_b8

    v8f acc = {};
#pragma unroll
    for (int ks = 0; ks < 18; ++ks) {
        v8h alo = *(const v8h*)(abase + ks * 32);
        v8h ahi = *(const v8h*)(abase + ks * 32 + 16);
        v16h am;
#pragma unroll
        for (int i = 0; i < 8; ++i) { am[i] = alo[i]; am[i + 8] = ahi[i]; }

        const _Float16* bp = bbase + ks * 128 * 32;   // [ks][n][kk] packed
        v8h blo = *(const v8h*)bp;
        v8h bhi = *(const v8h*)(bp + 16);
        v16h bm;
#pragma unroll
        for (int i = 0; i < 8; ++i) { bm[i] = blo[i]; bm[i + 8] = bhi[i]; }

        acc = __builtin_amdgcn_wmma_f32_16x16x32_f16(false, am, false, bm,
                                                     (short)0, acc, false, false);
    }
    float bias = bfc1[n];
#pragma unroll
    for (int r = 0; r < 8; ++r) {
        float val = fmaxf(acc[r] + bias, 0.f);
        h[(mtile * 16 + r + 8 * half) * 128 + n] = (_Float16)val;
    }
}

// ---------------------------------------------------------------------------
// FC2: [1024 x 128] x [128 x 10] + bias, f32 out (2.6 MFLOP -> scalar VALU)
// ---------------------------------------------------------------------------
__global__ void k_fc2(const _Float16* __restrict__ h, const float* __restrict__ wfc2,
                      const float* __restrict__ bfc2, float* __restrict__ out)
{
    int idx = blockIdx.x * blockDim.x + threadIdx.x;
    if (idx >= 1024 * 10) return;
    int o = idx % 10, b = idx / 10;
    const _Float16* hp = h + b * 128;
    const float* wp = wfc2 + o * 128;
    float s = bfc2[o];
#pragma unroll 8
    for (int k = 0; k < 128; ++k) s += (float)hp[k] * wp[k];
    out[idx] = s;
}

extern "C" void kernel_launch(void* const* d_in, const int* in_sizes, int n_in,
                              void* d_out, int out_size, void* d_ws, size_t ws_size,
                              hipStream_t stream)
{
    const int*   x_img = (const int*)d_in[0];
    const int*   y_int = (const int*)d_in[1];
    const float* bias1 = (const float*)d_in[2];
    const float* g1 = (const float*)d_in[3];
    const float* b1 = (const float*)d_in[4];
    const float* m1 = (const float*)d_in[5];
    const float* v1 = (const float*)d_in[6];
    const float* g2 = (const float*)d_in[7];
    const float* b2 = (const float*)d_in[8];
    const float* m2 = (const float*)d_in[9];
    const float* v2 = (const float*)d_in[10];
    const float* g3 = (const float*)d_in[11];
    const float* b3 = (const float*)d_in[12];
    const float* m3 = (const float*)d_in[13];
    const float* v3 = (const float*)d_in[14];
    const float* w2 = (const float*)d_in[15];
    const float* bias2 = (const float*)d_in[16];
    const float* w3 = (const float*)d_in[17];
    const float* bias3 = (const float*)d_in[18];
    const float* w_fc1 = (const float*)d_in[19];
    const float* b_fc1 = (const float*)d_in[20];
    const float* w_fc2 = (const float*)d_in[21];
    const float* b_fc2 = (const float*)d_in[22];

    char* ws = (char*)d_ws;
    _Float16* feat1p = (_Float16*)(ws + FEAT1P_OFF);
    _Float16* feat2p = (_Float16*)(ws + FEAT2P_OFF);
    _Float16* feat3  = (_Float16*)(ws + FEAT3_OFF);
    _Float16* wf1s   = (_Float16*)(ws + WF1S_OFF);
    _Float16* hbuf   = (_Float16*)(ws + HBUF_OFF);

    // zero halo'd activation buffers: 856,064 x 16B = 3344 blocks x 256
    k_zero<<<3344, 256, 0, stream>>>((v4u*)ws, ZERO_N16);
    // conv1: 1024*196 = 200,704 threads = 784 x 256
    k_conv1_bn_pool<<<784, 256, 0, stream>>>(x_img, y_int, bias1, g1, b1, m1, v1, feat1p);
    // FC1 weight pack
    k_wcvt<<<288, 256, 0, stream>>>(w_fc1, wf1s);
    // conv2: 64 imgtiles * 49 quads * 2 ntiles = 6272 waves = 784 blocks x 8 waves
    k_conv2_bn_pool<<<784, 256, 0, stream>>>(feat1p, w2, bias2, g2, b2, m2, v2, feat2p);
    // conv3: 64 * 9 * 4 = 2304 waves = 288 blocks x 8 waves
    k_conv3_bn_pool<<<288, 256, 0, stream>>>(feat2p, w3, bias3, g3, b3, m3, v3, feat3);
    // fc1: 64 mtiles * 8 ntiles = 512 waves = 64 blocks x 8 waves
    k_fc1<<<64, 256, 0, stream>>>(feat3, wf1s, b_fc1, hbuf);
    // fc2: 10240 threads
    k_fc2<<<40, 256, 0, stream>>>(hbuf, w_fc2, b_fc2, (float*)d_out);
}
